// LSTM_27281632264363
// MI455X (gfx1250) — matrix operand order repaired and checked by reading, not compiled, not run on hardware
//
#include <hip/hip_runtime.h>
#include <hip/hip_bf16.h>
#include <math.h>
#include <stdint.h>

#define B_    256
#define T_    256
#define I_    256
#define H_    1024
#define FOURH 4096
#define KTOT  1280           // I + H
#define NKC   (KTOT / 32)    // 40 K-chunks
#define TAU_  0.05f
#define MHU_  1.5f

typedef __attribute__((ext_vector_type(16))) __bf16 v16bf;
typedef __attribute__((ext_vector_type(8)))  float  v8f;

__device__ __forceinline__ unsigned short f2bf(float f) {
  unsigned int u = __float_as_uint(f);
  u += 0x7FFFu + ((u >> 16) & 1u);       // round-to-nearest-even
  return (unsigned short)(u >> 16);
}

// Build W_T[n][k] (bf16), n in [0,4096), k in [0,1280): rows 0..255 = Wx, 256..1279 = Wh.
__global__ void convert_weights(const float* __restrict__ Wx,
                                const float* __restrict__ Wh,
                                unsigned short* __restrict__ Wt) {
  int idx = blockIdx.x * blockDim.x + threadIdx.x;   // 4096*1280 total
  int n = idx / KTOT;
  int k = idx - n * KTOT;
  float v = (k < I_) ? Wx[(size_t)k * FOURH + n]
                     : Wh[(size_t)(k - I_) * FOURH + n];
  Wt[idx] = f2bf(v);
}

// ---- staging helpers -------------------------------------------------------

// A tile: 64 rows x 32 K, f32 -> bf16 with ODE substitution on h[:, :2].
__device__ __forceinline__ void stage_A(unsigned short* Albuf, int kc,
                                        const float* __restrict__ x,
                                        const float* __restrict__ h_src,
                                        int h_stride, int apply_ode, int t,
                                        int rowBlock, int ar, int aks) {
  const int k0 = kc * 32;
  float f[8];
  if (k0 < I_) {
    const float* src = x + ((size_t)(rowBlock + ar) * T_ + t) * I_ + k0 + aks;
    #pragma unroll
    for (int j = 0; j < 8; ++j) f[j] = src[j];
  } else {
    const int hc = k0 - I_ + aks;
    const float* src = h_src + (size_t)(rowBlock + ar) * h_stride + hc;
    #pragma unroll
    for (int j = 0; j < 8; ++j) f[j] = src[j];
    if (apply_ode && hc == 0) {
      float x0 = f[0], x1 = f[1];
      f[0] = x0 + TAU_ * (MHU_ * x0 + x1 * (1.0f / MHU_));
      f[1] = x1 - TAU_ * (MHU_ * x0);
    }
  }
  unsigned int p[4];
  #pragma unroll
  for (int j = 0; j < 4; ++j)
    p[j] = (unsigned)f2bf(f[2 * j]) | ((unsigned)f2bf(f[2 * j + 1]) << 16);
  *(uint4*)&Albuf[ar * 32 + aks] = make_uint4(p[0], p[1], p[2], p[3]);
}

// B tiles: 4 gates x 32 n x 32 K bf16, copied global->LDS with the CDNA5
// async-to-LDS path (ASYNCcnt).  32 bytes per thread = 2x b128.
__device__ __forceinline__ void stage_B_async(unsigned short* Blbuf, int kc,
                                              const unsigned short* __restrict__ Wt,
                                              size_t wt_row, int brow, int bhalf) {
  const int k0 = kc * 32;
  // Flat address of __shared__ = LDS aperture | group-relative offset; the
  // low 32 bits are the LDS byte offset the async op wants in VDST.
  unsigned lds = (unsigned)(uintptr_t)&Blbuf[brow * 32 + bhalf * 16];
  unsigned long long ga =
      (unsigned long long)(uintptr_t)(Wt + wt_row + k0 + bhalf * 16);
  asm volatile(
      "global_load_async_to_lds_b128 %0, %1, off\n\t"
      "global_load_async_to_lds_b128 %0, %1, off offset:16"
      :: "v"(lds), "v"(ga) : "memory");
}

// ---- one timestep ----------------------------------------------------------
// Block = 256 threads (8 waves). Tile = 64 batch rows x 32 h-cols (4 gate tiles).
__global__ __launch_bounds__(256)
void lstm_step(const float* __restrict__ x,
               const float* __restrict__ h_src, int h_stride, int apply_ode,
               const unsigned short* __restrict__ Wt,
               const float* __restrict__ bias,
               const float* __restrict__ c_src,
               float* __restrict__ c_dst,
               float* __restrict__ out, int t)
{
  __shared__ __align__(16) unsigned short Al[2][64 * 32];    // A [buf][row][k]
  __shared__ __align__(16) unsigned short Bl[2][128 * 32];   // B [buf][(g*32+n)][k]
  __shared__ __align__(16) float gbuf[4 * 64 * 32];          // gate results

  const int tid  = threadIdx.x;
  const int lane = tid & 31;
  const int wid  = tid >> 5;
  const int g    = wid >> 1;        // gate owned by this wave
  const int rh   = wid & 1;         // row-half (32 rows)
  const int rowBlock = blockIdx.x * 64;   // batch rows
  const int colBlock = blockIdx.y * 32;   // h columns

  v8f acc[2][2];
  {
    v8f z = {};
    acc[0][0] = z; acc[0][1] = z; acc[1][0] = z; acc[1][1] = z;
  }

  // staging roles
  const int ar  = tid >> 2;               // A row 0..63
  const int aks = (tid & 3) << 3;         // A k-start: 0,8,16,24
  const int brow  = tid >> 1;             // B row 0..127 = g*32+n
  const int bhalf = tid & 1;
  const int bg = brow >> 5;
  const int bn = brow & 31;
  const size_t wt_row = (size_t)(bg * H_ + colBlock + bn) * KTOT;

  // prologue: stage chunk 0 into buffer 0
  stage_A(Al[0], 0, x, h_src, h_stride, apply_ode, t, rowBlock, ar, aks);
  stage_B_async(Bl[0], 0, Wt, wt_row, brow, bhalf);

  for (int kc = 0; kc < NKC; ++kc) {
    const int cur = kc & 1;
    // drain this wave's async B copies, then join the workgroup (includes DS fence)
    asm volatile("s_wait_asynccnt 0x0" ::: "memory");
    __syncthreads();

    // overlap: issue staging of chunk kc+1 into the other buffer
    if (kc + 1 < NKC) {
      stage_A(Al[cur ^ 1], kc + 1, x, h_src, h_stride, apply_ode, t,
              rowBlock, ar, aks);
      stage_B_async(Bl[cur ^ 1], kc + 1, Wt, wt_row, brow, bhalf);
    }

    // ---- fragments + WMMA on current buffer ----
    union Frag { v16bf v; uint4 q[2]; } a[2], bfrag[2];
    const int half = lane >> 4;
    const int l16  = lane & 15;
    #pragma unroll
    for (int rt = 0; rt < 2; ++rt) {
      const int row = rh * 32 + rt * 16 + l16;
      // A layout: lanes 0-15 hold K 0..7 & 16..23; lanes 16-31 hold K 8..15 & 24..31
      a[rt].q[0] = *(const uint4*)&Al[cur][row * 32 + 8 * half];
      a[rt].q[1] = *(const uint4*)&Al[cur][row * 32 + 16 + 8 * half];
    }
    #pragma unroll
    for (int ct = 0; ct < 2; ++ct) {
      const int nrow = g * 32 + ct * 16 + l16;
      // B layout: lanes 0-15 hold K 0..7 & 16..23; lanes 16-31 hold K 8..15 & 24..31
      bfrag[ct].q[0] = *(const uint4*)&Bl[cur][nrow * 32 + 8 * half];
      bfrag[ct].q[1] = *(const uint4*)&Bl[cur][nrow * 32 + 16 + 8 * half];
    }
    #pragma unroll
    for (int rt = 0; rt < 2; ++rt)
      #pragma unroll
      for (int ct = 0; ct < 2; ++ct)
        acc[rt][ct] = __builtin_amdgcn_wmma_f32_16x16x32_bf16(
            false, a[rt].v, false, bfrag[ct].v, (short)0, acc[rt][ct], false, false);
  }

  __syncthreads();
  // scatter accumulators to gbuf (C layout: VGPR v -> M = v + 8*(lane/16), N = lane%16)
  {
    const int half = lane >> 4;
    const int l16  = lane & 15;
    #pragma unroll
    for (int rt = 0; rt < 2; ++rt)
      #pragma unroll
      for (int ct = 0; ct < 2; ++ct)
        #pragma unroll
        for (int v = 0; v < 8; ++v) {
          int row = rh * 32 + rt * 16 + v + 8 * half;
          int col = ct * 16 + l16;
          gbuf[(g * 64 + row) * 32 + col] = acc[rt][ct][v];
        }
  }
  __syncthreads();

  // elementwise LSTM update: 64*32 = 2048 elements, 8 per thread
  #pragma unroll
  for (int j = 0; j < 8; ++j) {
    int pos = tid * 8 + j;
    int r  = pos >> 5;
    int cc = pos & 31;
    int b   = rowBlock + r;
    int col = colBlock + cc;
    float gi = gbuf[(0 * 64 + r) * 32 + cc] + bias[0 * H_ + col];
    float gf = gbuf[(1 * 64 + r) * 32 + cc] + bias[1 * H_ + col];
    float gg = gbuf[(2 * 64 + r) * 32 + cc] + bias[2 * H_ + col];
    float go = gbuf[(3 * 64 + r) * 32 + cc] + bias[3 * H_ + col];
    float si = 1.f / (1.f + __expf(-gi));
    float sf = 1.f / (1.f + __expf(-gf));
    float so = 1.f / (1.f + __expf(-go));
    float tg = tanhf(gg);
    float cold = c_src[(size_t)b * H_ + col];
    float cn = sf * cold + si * tg;
    float hn = so * tanhf(cn);
    c_dst[(size_t)b * H_ + col] = cn;
    out[((size_t)b * T_ + t) * H_ + col] = hn;
  }
}

__global__ void finalize_k(const float* __restrict__ out_main,
                           const float* __restrict__ c_ws,
                           float* __restrict__ hN, float* __restrict__ cN) {
  int i = blockIdx.x * blockDim.x + threadIdx.x;  // B*H
  int b = i >> 10;            // / H
  int col = i & (H_ - 1);
  hN[i] = out_main[((size_t)b * T_ + (T_ - 1)) * H_ + col];
  cN[i] = c_ws[i];
}

extern "C" void kernel_launch(void* const* d_in, const int* in_sizes, int n_in,
                              void* d_out, int out_size, void* d_ws, size_t ws_size,
                              hipStream_t stream) {
  const float* x    = (const float*)d_in[0];   // (B,T,I)
  const float* h0   = (const float*)d_in[1];   // (1,B,H)
  const float* c0   = (const float*)d_in[2];   // (1,B,H)
  const float* Wx   = (const float*)d_in[3];   // (I,4H)
  const float* Wh   = (const float*)d_in[4];   // (H,4H)
  const float* bias = (const float*)d_in[5];   // (4H,)
  float* out = (float*)d_out;

  unsigned short* Wt = (unsigned short*)d_ws;                       // 10 MB bf16 weights
  float* c_ws = (float*)((char*)d_ws +
                         (size_t)FOURH * KTOT * sizeof(unsigned short)); // 1 MB c state

  convert_weights<<<(FOURH * KTOT) / 256, 256, 0, stream>>>(Wx, Wh, Wt);

  dim3 grid(B_ / 64, H_ / 32);
  for (int t = 0; t < T_; ++t) {
    const float* h_src = (t == 0) ? h0 : (out + (size_t)(t - 1) * H_);
    int h_stride       = (t == 0) ? H_ : T_ * H_;
    const float* c_src = (t == 0) ? c0 : (const float*)c_ws;
    lstm_step<<<grid, 256, 0, stream>>>(x, h_src, h_stride, (t > 0) ? 1 : 0,
                                        Wt, bias, c_src, c_ws, out, t);
  }

  float* hN = out + (size_t)B_ * T_ * H_;
  float* cN = hN + (size_t)B_ * H_;
  finalize_k<<<(B_ * H_) / 256, 256, 0, stream>>>(out, c_ws, hN, cN);
}
